// Uni3DPointcloudEncoder_59914793779537
// MI455X (gfx1250) — matrix operand order repaired and checked by reading, not compiled, not run on hardware
//
#include <hip/hip_runtime.h>
#include <math.h>

#define B_    8
#define N_    8192
#define G_    512
#define K_    32
#define ENC_  512
#define D_    1024
#define L_    4
#define H_    16
#define DH_   64
#define DFF_  4096
#define S_    513
#define R_    (B_*S_)          // 4104 transformer rows
#define ROWS_ (B_*G_*K_)       // 131072 pointnet rows

typedef __attribute__((ext_vector_type(16))) __bf16 bf16x16;
typedef __attribute__((ext_vector_type(8)))  float  f32x8;
typedef __attribute__((ext_vector_type(4)))  int    v4i;

union Frag { unsigned short u[16]; bf16x16 v; };

__device__ __forceinline__ unsigned short f2bf(float f){
  unsigned int u = __float_as_uint(f);
  u += 0x7fffu + ((u >> 16) & 1u);           // round-to-nearest-even
  return (unsigned short)(u >> 16);
}
__device__ __forceinline__ float gelu_f(float x){
  return 0.5f * x * (1.0f + erff(x * 0.70710678118654752440f));
}
__device__ __forceinline__ void atomicMaxF(float* a, float v){
  if (v >= 0.0f) atomicMax((int*)a, __float_as_int(v));
  else           atomicMin((unsigned int*)a, __float_as_uint(v));
}

// ---- CDNA5 async global->LDS copy (guarded; falls back to plain LDS stores) ----
#if defined(__gfx1250__) && __has_builtin(__builtin_amdgcn_global_load_async_to_lds_b128)
#define HAVE_ASYNC 1
#else
#define HAVE_ASYNC 0
#endif

typedef __attribute__((address_space(1))) v4i GV4;   // global int4
typedef __attribute__((address_space(3))) v4i LV4;   // LDS int4

__device__ __forceinline__ void cp16_g2l(void* lds, const void* g){
#if HAVE_ASYNC
  __builtin_amdgcn_global_load_async_to_lds_b128((GV4*)g, (LV4*)lds, 0, 0);
#else
  *(uint4*)lds = *(const uint4*)g;
#endif
}
__device__ __forceinline__ void wait_async0(){
#if HAVE_ASYNC
 #if __has_builtin(__builtin_amdgcn_s_wait_asynccnt)
  __builtin_amdgcn_s_wait_asynccnt(0);
 #else
  asm volatile("s_wait_asynccnt 0x0" ::: "memory");
 #endif
#endif
}

// ---------------------------------------------------------------- FPS
__global__ void fps_kernel(const float* __restrict__ pcd, int* __restrict__ fidx,
                           float* __restrict__ dists){
  int b = blockIdx.x, t = threadIdx.x;
  __shared__ float sv[256]; __shared__ int si[256];
  __shared__ int cur; __shared__ float cx, cy, cz;
  for (int n = t; n < N_; n += 256) dists[(size_t)b*N_ + n] = 1e10f;
  if (t == 0) cur = 0;
  __syncthreads();
  for (int step = 0; step < G_; ++step){
    if (t == 0){
      fidx[b*G_ + step] = cur;
      const float* p = pcd + ((size_t)b*N_ + cur)*6;
      cx = p[0]; cy = p[1]; cz = p[2];
    }
    __syncthreads();
    float bm = -1.0f; int bi = 0;
    for (int n = t; n < N_; n += 256){
      const float* p = pcd + ((size_t)b*N_ + n)*6;
      float dx = p[0]-cx, dy = p[1]-cy, dz = p[2]-cz;
      float d  = dx*dx + dy*dy + dz*dz;
      float od = dists[(size_t)b*N_ + n];
      float nd = od < d ? od : d;
      dists[(size_t)b*N_ + n] = nd;
      if (nd > bm){ bm = nd; bi = n; }
    }
    sv[t] = bm; si[t] = bi; __syncthreads();
    for (int off = 128; off > 0; off >>= 1){
      if (t < off){
        if (sv[t+off] > sv[t] || (sv[t+off] == sv[t] && si[t+off] < si[t])){
          sv[t] = sv[t+off]; si[t] = si[t+off];
        }
      }
      __syncthreads();
    }
    if (t == 0) cur = si[0];
    __syncthreads();
  }
}

// ---------------------------------------------------------------- kNN (K smallest)
__global__ void knn_kernel(const float* __restrict__ pcd, const int* __restrict__ fidx,
                           int* __restrict__ gidx){
  int g = blockIdx.x, b = blockIdx.y, t = threadIdx.x;
  __shared__ float d[N_];
  __shared__ float sv[256]; __shared__ int si[256];
  int ci = fidx[b*G_ + g];
  const float* cp = pcd + ((size_t)b*N_ + ci)*6;
  float cx = cp[0], cy = cp[1], cz = cp[2];
  float cc = cx*cx + cy*cy + cz*cz;
  for (int n = t; n < N_; n += 256){
    const float* p = pcd + ((size_t)b*N_ + n)*6;
    float xx  = p[0]*p[0] + p[1]*p[1] + p[2]*p[2];
    float dot = cx*p[0] + cy*p[1] + cz*p[2];
    d[n] = cc + xx - 2.0f*dot;
  }
  __syncthreads();
  for (int kIt = 0; kIt < K_; ++kIt){
    float bm = 1e30f; int bi = 0;
    for (int n = t; n < N_; n += 256){ float v = d[n]; if (v < bm){ bm = v; bi = n; } }
    sv[t] = bm; si[t] = bi; __syncthreads();
    for (int off = 128; off > 0; off >>= 1){
      if (t < off){
        if (sv[t+off] < sv[t] || (sv[t+off] == sv[t] && si[t+off] < si[t])){
          sv[t] = sv[t+off]; si[t] = si[t+off];
        }
      }
      __syncthreads();
    }
    if (t == 0){ gidx[((size_t)b*G_ + g)*K_ + kIt] = si[0]; d[si[0]] = 1e30f; }
    __syncthreads();
  }
}

// ---------------------------------------------------------------- gather neighborhood feats
__global__ void gather_kernel(const float* __restrict__ pcd, const int* __restrict__ fidx,
                              const int* __restrict__ gidx, float* __restrict__ feats,
                              float* __restrict__ center){
  size_t i = (size_t)blockIdx.x*blockDim.x + threadIdx.x;
  if (i >= (size_t)ROWS_) return;
  int k = (int)(i % K_); size_t gg = i / K_;
  int g = (int)(gg % G_); int b = (int)(gg / G_);
  int ci = fidx[b*G_ + g];
  const float* cp = pcd + ((size_t)b*N_ + ci)*6;
  int j = gidx[i];
  const float* p = pcd + ((size_t)b*N_ + j)*6;
  float* f = feats + i*6;
  f[0] = p[0]-cp[0]; f[1] = p[1]-cp[1]; f[2] = p[2]-cp[2];
  f[3] = p[3];       f[4] = p[4];       f[5] = p[5];
  if (k == 0){ float* c = center + gg*3; c[0]=cp[0]; c[1]=cp[1]; c[2]=cp[2]; }
}

// ---------------------------------------------------------------- tiny scalar GEMMs
__global__ void conv1_kernel(const float* __restrict__ feats, const float* __restrict__ w,
                             const float* __restrict__ bias, float* __restrict__ out){
  size_t i = (size_t)blockIdx.x*blockDim.x + threadIdx.x;
  if (i >= (size_t)ROWS_*128) return;
  int c = (int)(i & 127); size_t r = i >> 7;
  const float* fr = feats + r*6;
  float a = bias[c];
#pragma unroll
  for (int k = 0; k < 6; ++k) a += fr[k]*w[k*128 + c];
  out[i] = a;
}

__global__ void pos1_kernel(const float* __restrict__ center, const float* __restrict__ w,
                            const float* __restrict__ bias, float* __restrict__ out){
  size_t i = (size_t)blockIdx.x*blockDim.x + threadIdx.x;
  if (i >= (size_t)(B_*G_)*128) return;
  int c = (int)(i & 127); size_t r = i >> 7;
  const float* cr = center + r*3;
  float a = bias[c];
#pragma unroll
  for (int k = 0; k < 3; ++k) a += cr[k]*w[k*128 + c];
  out[i] = gelu_f(a);
}

// ---------------------------------------------------------------- utility
__global__ void fill_kernel(float* __restrict__ p, float v, size_t n){
  size_t i = (size_t)blockIdx.x*blockDim.x + threadIdx.x;
  if (i < n) p[i] = v;
}

__global__ void tobf16_kernel(const float* __restrict__ src, unsigned short* __restrict__ dst, size_t n){
  size_t i = (size_t)blockIdx.x*blockDim.x + threadIdx.x;
  if (i < n) dst[i] = f2bf(src[i]);
}

// transpose+convert weights: W[K][N] f32 -> Wt[N][K] bf16
__global__ void wtrans_kernel(const float* __restrict__ W, unsigned short* __restrict__ Wt,
                              int K, int N){
  __shared__ unsigned short tile[32][33];
  int kt = blockIdx.y*32, nt = blockIdx.x*32;
  int t = threadIdx.x;
  int tr = t >> 5, tc = t & 31;
  for (int rr = tr; rr < 32; rr += 8){
    int k = kt + rr, n = nt + tc;
    tile[rr][tc] = (k < K && n < N) ? f2bf(W[(size_t)k*N + n]) : (unsigned short)0;
  }
  __syncthreads();
  for (int rr = tr; rr < 32; rr += 8){
    int n = nt + rr, k = kt + tc;
    if (n < N && k < K) Wt[(size_t)n*K + k] = tile[tc][rr];
  }
}

// ---------------------------------------------------------------- BatchNorm (stats over all rows)
__global__ void bn_stats_kernel(const float* __restrict__ x, float* __restrict__ sums,
                                int rows, int C){
  int r0 = blockIdx.x * 256;
  int rend = r0 + 256 < rows ? r0 + 256 : rows;
  for (int c = threadIdx.x; c < C; c += 256){
    float s = 0.f, s2 = 0.f;
    for (int r = r0; r < rend; ++r){
      float v = x[(size_t)r*C + c];
      s += v; s2 += v*v;
    }
    atomicAdd(&sums[c], s);
    atomicAdd(&sums[C + c], s2);
  }
}

__global__ void bn_apply_relu_kernel(float* __restrict__ x, const float* __restrict__ sums,
                                     const float* __restrict__ g, const float* __restrict__ b,
                                     size_t rows, int C){
  size_t i = (size_t)blockIdx.x*blockDim.x + threadIdx.x;
  size_t total = rows * (size_t)C;
  if (i >= total) return;
  int c = (int)(i % C);
  float inv  = 1.0f / (float)rows;
  float mean = sums[c]*inv;
  float var  = sums[C + c]*inv - mean*mean;
  float sc   = g[c] * rsqrtf(var + 1e-5f);
  float sh   = b[c] - mean*sc;
  x[i] = fmaxf(x[i]*sc + sh, 0.0f);
}

// ---------------------------------------------------------------- max over K (fg)
__global__ void rowmax_kernel(const float* __restrict__ f2, float* __restrict__ fg){
  size_t i = (size_t)blockIdx.x*blockDim.x + threadIdx.x;
  if (i >= (size_t)(B_*G_)*256) return;
  int c = (int)(i & 255); size_t gg = i >> 8;
  float m = -1e30f;
#pragma unroll 4
  for (int k = 0; k < K_; ++k) m = fmaxf(m, f2[(gg*K_ + k)*256 + c]);
  fg[i] = m;
}

// ---------------------------------------------------------------- bf16 WMMA GEMM
// A (bf16, MxK row-major), Wt (bf16, NxK row-major i.e. W transposed).
// Double-buffered LDS staged via async global->LDS b128 copies.
// concatSplit!=0: A column k<split comes from A2[(m/32)*split + k], else A[m*(K-split)+(k-split)]
// poolGroup!=0 : atomic max into C[(m/poolGroup)*N + n] (fused max-pool epilogue)
__global__ __launch_bounds__(128)
void gemm_bf16_kernel(const unsigned short* __restrict__ A, const unsigned short* __restrict__ A2,
                      const unsigned short* __restrict__ Wt, const float* __restrict__ bias,
                      const float* __restrict__ residual, float* __restrict__ C,
                      int M, int N, int K, int act, int concatSplit, int poolGroup){
  __shared__ __align__(16) unsigned short As[2][64][40];
  __shared__ __align__(16) unsigned short Bs[2][64][40];
  int m0 = blockIdx.y*64, n0 = blockIdx.x*64;
  int tid = threadIdx.x, lane = tid & 31, wid = tid >> 5;
  int waveM = (wid >> 1)*32, waveN = (wid & 1)*32;
  int l = lane & 15, g = lane >> 4;

  auto stage = [&](int buf, int kt){
    // A tile: 64 rows x 32 bf16 = 256 x 16B chunks
    for (int ch = tid; ch < 256; ch += 128){
      int r = ch >> 2, c = ch & 3;
      int m = m0 + r, ke = kt + c*8;
      if (m < M){
        const unsigned short* src;
        if (concatSplit && ke < concatSplit) src = A2 + (size_t)(m >> 5)*concatSplit + ke;
        else if (concatSplit)                src = A  + (size_t)m*(K - concatSplit) + (ke - concatSplit);
        else                                 src = A  + (size_t)m*K + ke;
        cp16_g2l(&As[buf][r][c*8], src);
        if (kt + 64 <= K) __builtin_prefetch(src + 32, 0, 0);
      }
    }
    // B tile: 64 n-rows x 32 bf16
    for (int ch = tid; ch < 256; ch += 128){
      int r = ch >> 2, c = ch & 3;
      int n = n0 + r;
      if (n < N){
        const unsigned short* src = Wt + (size_t)n*K + kt + c*8;
        cp16_g2l(&Bs[buf][r][c*8], src);
        if (kt + 64 <= K) __builtin_prefetch(src + 32, 0, 0);
      }
    }
  };

  f32x8 acc[2][2] = {};
  stage(0, 0);
  wait_async0();
  __syncthreads();
  int buf = 0;
  for (int kt = 0; kt < K; kt += 32){
    if (kt + 32 < K) stage(buf ^ 1, kt + 32);

    bf16x16 afr[2], bfr[2];
#pragma unroll
    for (int mi = 0; mi < 2; ++mi){
      Frag f; int row = waveM + mi*16 + l;
#pragma unroll
      for (int j = 0; j < 8; ++j){
        f.u[j]     = As[buf][row][g*8 + j];        // K 0..7 / 8..15
        f.u[8 + j] = As[buf][row][16 + g*8 + j];   // K 16..23 / 24..31
      }
      afr[mi] = f.v;
    }
#pragma unroll
    for (int ni = 0; ni < 2; ++ni){
      Frag f; int col = waveN + ni*16 + l;
#pragma unroll
      for (int j = 0; j < 16; ++j) f.u[j] = Bs[buf][col][(j < 8) ? (g*8 + j) : (16 + g*8 + (j - 8))]; // K 0..7,16..23 / 8..15,24..31
      bfr[ni] = f.v;
    }
#pragma unroll
    for (int mi = 0; mi < 2; ++mi)
#pragma unroll
      for (int ni = 0; ni < 2; ++ni)
        acc[mi][ni] = __builtin_amdgcn_wmma_f32_16x16x32_bf16(
            false, afr[mi], false, bfr[ni], (short)0, acc[mi][ni], false, false);

    wait_async0();
    __syncthreads();
    buf ^= 1;
  }

  // epilogue
#pragma unroll
  for (int mi = 0; mi < 2; ++mi){
#pragma unroll
    for (int ni = 0; ni < 2; ++ni){
      int n = n0 + waveN + ni*16 + l;
      if (n >= N) continue;
      float bv = bias ? bias[n] : 0.0f;
#pragma unroll
      for (int r = 0; r < 8; ++r){
        int m = m0 + waveM + mi*16 + r + 8*g;
        if (m >= M) continue;
        float v = acc[mi][ni][r] + bv;
        if (act == 1)      v = fmaxf(v, 0.0f);
        else if (act == 2) v = gelu_f(v);
        if (residual) v += residual[(size_t)m*N + n];
        if (poolGroup) atomicMaxF(&C[(size_t)(m / poolGroup)*N + n], v);
        else           C[(size_t)m*N + n] = v;
      }
    }
  }
}

// ---------------------------------------------------------------- LayerNorm (row-wise)
__global__ void ln_kernel(const float* __restrict__ x, float* __restrict__ out,
                          const float* __restrict__ g, const float* __restrict__ b, int D){
  int row = blockIdx.x, t = threadIdx.x;
  const float* xr = x + (size_t)row*D;
  __shared__ float s1[256], s2[256];
  float a = 0.f, a2 = 0.f;
  for (int c = t; c < D; c += 256){ float v = xr[c]; a += v; a2 += v*v; }
  s1[t] = a; s2[t] = a2; __syncthreads();
  for (int off = 128; off > 0; off >>= 1){
    if (t < off){ s1[t] += s1[t+off]; s2[t] += s2[t+off]; }
    __syncthreads();
  }
  float mu = s1[0]/D, var = s2[0]/D - mu*mu, iv = rsqrtf(var + 1e-5f);
  for (int c = t; c < D; c += 256)
    out[(size_t)row*D + c] = (xr[c]-mu)*iv*g[c] + b[c];
}

// ---------------------------------------------------------------- Flash attention (WMMA, bf16 QKV)
// grid: (ceil(S/16), H, B); block: 1 wave.
__global__ __launch_bounds__(32)
void attn_kernel(const unsigned short* __restrict__ qkv, float* __restrict__ o){
  const int qt = blockIdx.x, h = blockIdx.y, b = blockIdx.z;
  const int lane = threadIdx.x, l = lane & 15, g = lane >> 4;
  __shared__ float S_lds[16][33];
  __shared__ __align__(16) unsigned short P_lds[16][32];
  __shared__ float rm[16], rl[16], alf[16];
  const size_t rs = 3*D_;
  const unsigned short* Q  = qkv + (size_t)h*DH_;
  const unsigned short* Kp = qkv + D_   + (size_t)h*DH_;
  const unsigned short* Vp = qkv + 2*D_ + (size_t)h*DH_;

  // Q A-fragments (two 32-wide K chunks of DH=64) as b128 loads
  bf16x16 qf[2];
  {
    int s = qt*16 + l; bool ok = s < S_;
    const unsigned short* qr = Q + ((size_t)b*S_ + (ok ? s : 0))*rs;
#pragma unroll
    for (int c = 0; c < 2; ++c){
      Frag f;
      if (ok){
        *(uint4*)&f.u[0] = *(const uint4*)(qr + c*32 + g*8);
        *(uint4*)&f.u[8] = *(const uint4*)(qr + c*32 + 16 + g*8);
      } else {
#pragma unroll
        for (int j = 0; j < 16; ++j) f.u[j] = 0;
      }
      qf[c] = f.v;
    }
  }
  if (lane < 16){ rm[lane] = -1e30f; rl[lane] = 0.0f; }
  f32x8 oacc[4] = {};
  __syncthreads();

  const int nkt = (S_ + 31) / 32;   // 17
  for (int kk = 0; kk < nkt; ++kk){
    int kb = kk*32;
    // scores: two 16x16 tiles -> S_lds[16][32]
#pragma unroll
    for (int t2 = 0; t2 < 2; ++t2){
      f32x8 sacc = {};
      int kr = kb + t2*16 + l;          // this lane's key row (B-frag column)
      bool kok = kr < S_;
      const unsigned short* krow = Kp + ((size_t)b*S_ + (kok ? kr : 0))*rs;
#pragma unroll
      for (int c = 0; c < 2; ++c){
        Frag f;
        if (kok){
          *(uint4*)&f.u[0] = *(const uint4*)(krow + c*32 + g*8);
          *(uint4*)&f.u[8] = *(const uint4*)(krow + c*32 + 16 + g*8);
        } else {
#pragma unroll
          for (int j = 0; j < 16; ++j) f.u[j] = 0;
        }
        sacc = __builtin_amdgcn_wmma_f32_16x16x32_bf16(
            false, qf[c], false, f.v, (short)0, sacc, false, false);
      }
#pragma unroll
      for (int r = 0; r < 8; ++r){
        float v = sacc[r] * 0.125f;     // 1/sqrt(DH)
        if (!kok) v = -1e30f;
        S_lds[r + 8*g][t2*16 + l] = v;
      }
    }
    __syncthreads();
    if (lane < 16){
      int m = lane;
      float mx = rm[m];
#pragma unroll 4
      for (int j = 0; j < 32; ++j) mx = fmaxf(mx, S_lds[m][j]);
      float a = __expf(rm[m] - mx);
      float s = 0.f;
#pragma unroll 4
      for (int j = 0; j < 32; ++j){
        float e = __expf(S_lds[m][j] - mx);
        P_lds[m][j] = f2bf(e);
        s += e;
      }
      rl[m] = rl[m]*a + s; rm[m] = mx; alf[m] = a;
    }
    __syncthreads();
#pragma unroll
    for (int ni = 0; ni < 4; ++ni)
#pragma unroll
      for (int r = 0; r < 8; ++r) oacc[ni][r] *= alf[r + 8*g];
    // P A-fragment (16x32)
    Frag pf;
#pragma unroll
    for (int j = 0; j < 8; ++j){
      pf.u[j]     = P_lds[l][g*8 + j];
      pf.u[8 + j] = P_lds[l][16 + g*8 + j];
    }
    // V B-fragments (32 keys x 64 dh, 4 n-subtiles)
#pragma unroll
    for (int ni = 0; ni < 4; ++ni){
      Frag vf;
      int d = ni*16 + l;
#pragma unroll
      for (int j = 0; j < 16; ++j){
        int kr = kb + ((j < 8) ? (g*8 + j) : (16 + g*8 + (j - 8)));
        vf.u[j] = (kr < S_) ? Vp[((size_t)b*S_ + kr)*rs + d] : (unsigned short)0;
      }
      oacc[ni] = __builtin_amdgcn_wmma_f32_16x16x32_bf16(
          false, pf.v, false, vf.v, (short)0, oacc[ni], false, false);
    }
    __syncthreads();
  }
  // normalize + store
#pragma unroll
  for (int ni = 0; ni < 4; ++ni){
    int d = ni*16 + l;
#pragma unroll
    for (int r = 0; r < 8; ++r){
      int m = r + 8*g, s = qt*16 + m;
      if (s < S_)
        o[((size_t)b*S_ + s)*D_ + h*DH_ + d] = oacc[ni][r] / rl[m];
    }
  }
}

// ---------------------------------------------------------------- build sequence x = tokens+pos
__global__ void build_x_kernel(const float* __restrict__ xe, const float* __restrict__ posp,
                               const float* __restrict__ cls_tok, const float* __restrict__ cls_pos,
                               float* __restrict__ x){
  size_t i = (size_t)blockIdx.x*blockDim.x + threadIdx.x;
  if (i >= (size_t)R_*D_) return;
  int c = (int)(i % D_); size_t rsd = i / D_;
  int s = (int)(rsd % S_); int b = (int)(rsd / S_);
  float v;
  if (s == 0) v = cls_tok[c] + cls_pos[c];
  else {
    size_t idx = ((size_t)b*G_ + (s-1))*D_ + c;
    v = xe[idx] + posp[idx];
  }
  x[i] = v;
}

// ---------------------------------------------------------------- final LN + head
__global__ void final_head_kernel(const float* __restrict__ x, const float* __restrict__ g,
                                  const float* __restrict__ b, const float* __restrict__ w,
                                  const float* __restrict__ bias, float* __restrict__ out){
  int bi = blockIdx.x, t = threadIdx.x;
  const float* xr = x + (size_t)bi*S_*D_;   // row [:,0]
  __shared__ float s1[256], s2[256];
  __shared__ float hrow[D_];
  float a = 0.f, a2 = 0.f;
  for (int c = t; c < D_; c += 256){ float v = xr[c]; a += v; a2 += v*v; }
  s1[t] = a; s2[t] = a2; __syncthreads();
  for (int off = 128; off > 0; off >>= 1){
    if (t < off){ s1[t] += s1[t+off]; s2[t] += s2[t+off]; }
    __syncthreads();
  }
  float mu = s1[0]/D_, var = s2[0]/D_ - mu*mu, iv = rsqrtf(var + 1e-5f);
  for (int c = t; c < D_; c += 256) hrow[c] = (xr[c]-mu)*iv*g[c] + b[c];
  __syncthreads();
  for (int n = t; n < D_; n += 256){
    float acc = bias[n];
    for (int k = 0; k < D_; ++k) acc += hrow[k]*w[(size_t)k*D_ + n];
    out[(size_t)bi*D_ + n] = acc;
  }
}

// ---------------------------------------------------------------- host
static inline void gemmL(hipStream_t s, const unsigned short* A, const unsigned short* A2,
                         const unsigned short* Wt, const float* bias, const float* res, float* C,
                         int M, int N, int K, int act, int split, int pool){
  dim3 grid((N + 63)/64, (M + 63)/64);
  gemm_bf16_kernel<<<grid, 128, 0, s>>>(A, A2, Wt, bias, res, C, M, N, K, act, split, pool);
}
static inline int gs(size_t total){ return (int)((total + 255)/256); }
static inline void convA(hipStream_t s, const float* src, unsigned short* dst, size_t n){
  tobf16_kernel<<<gs(n), 256, 0, s>>>(src, dst, n);
}
static inline void convW(hipStream_t s, const float* W, unsigned short* Wt, int K, int N){
  wtrans_kernel<<<dim3((N+31)/32, (K+31)/32), 256, 0, s>>>(W, Wt, K, N);
}

extern "C" void kernel_launch(void* const* d_in, const int* in_sizes, int n_in,
                              void* d_out, int out_size, void* d_ws, size_t ws_size,
                              hipStream_t stream){
  (void)in_sizes; (void)n_in; (void)out_size; (void)ws_size;
  const float* pcd     = (const float*)d_in[0];
  const float* conv1_w = (const float*)d_in[1];
  const float* conv1_b = (const float*)d_in[2];
  const float* bn1_g   = (const float*)d_in[3];
  const float* bn1_b   = (const float*)d_in[4];
  const float* conv2_w = (const float*)d_in[5];
  const float* conv2_b = (const float*)d_in[6];
  const float* conv3_w = (const float*)d_in[7];
  const float* conv3_b = (const float*)d_in[8];
  const float* bn2_g   = (const float*)d_in[9];
  const float* bn2_b   = (const float*)d_in[10];
  const float* conv4_w = (const float*)d_in[11];
  const float* conv4_b = (const float*)d_in[12];
  const float* e2t_w   = (const float*)d_in[13];
  const float* e2t_b   = (const float*)d_in[14];
  const float* cls_tok = (const float*)d_in[15];
  const float* cls_pos = (const float*)d_in[16];
  const float* pos1_w  = (const float*)d_in[17];
  const float* pos1_b  = (const float*)d_in[18];
  const float* pos2_w  = (const float*)d_in[19];
  const float* pos2_b  = (const float*)d_in[20];
  const float* ln1_g   = (const float*)d_in[21];
  const float* ln1_b   = (const float*)d_in[22];
  const float* wqkv    = (const float*)d_in[23];
  const float* bqkv    = (const float*)d_in[24];
  const float* wo      = (const float*)d_in[25];
  const float* bo      = (const float*)d_in[26];
  const float* ln2_g   = (const float*)d_in[27];
  const float* ln2_b   = (const float*)d_in[28];
  const float* w1      = (const float*)d_in[29];
  const float* b1      = (const float*)d_in[30];
  const float* w2      = (const float*)d_in[31];
  const float* b2      = (const float*)d_in[32];
  const float* norm_g  = (const float*)d_in[33];
  const float* norm_b  = (const float*)d_in[34];
  const float* t2e_w   = (const float*)d_in[35];
  const float* t2e_b   = (const float*)d_in[36];
  float* out = (float*)d_out;

  // ---- workspace bump allocator (with lifetime-based aliasing) ----
  char* base = (char*)d_ws; size_t off = 0;
  auto alloc = [&](size_t bytes)->void*{
    void* p = base + off; off += bytes; off = (off + 255) & ~(size_t)255; return p;
  };
  int*   fidx   = (int*)  alloc((size_t)B_*G_*4);
  float* dists  = (float*)alloc((size_t)B_*N_*4);
  int*   gidx   = (int*)  alloc((size_t)B_*G_*K_*4);
  float* center = (float*)alloc((size_t)B_*G_*3*4);
  float* feats  = (float*)alloc((size_t)ROWS_*6*4);
  float* stats  = (float*)alloc(1024*4);
  float* tok    = (float*)alloc((size_t)B_*G_*ENC_*4);
  float* f1     = (float*)alloc((size_t)ROWS_*128*4);   // 64 MB
  float* f2     = (float*)alloc((size_t)ROWS_*256*4);   // 128 MB
  float* fg     = (float*)alloc((size_t)B_*G_*256*4);   // 4 MB
  float* f3     = (float*)alloc((size_t)ROWS_*512*4);   // 256 MB
  unsigned short* abuf  = (unsigned short*)alloc((size_t)ROWS_*512*2);  // 134 MB (max A panel, bf16)
  unsigned short* wtbuf = (unsigned short*)alloc((size_t)DFF_*D_*2);    // 8.4 MB (max Wt, bf16)
  unsigned short* fgbf  = (unsigned short*)alloc((size_t)B_*G_*256*2);  // 2 MB
  unsigned short* qkvbf = (unsigned short*)alloc((size_t)R_*3*D_*2);    // 25 MB
  // transformer arena: aliases f1+f2 region (dead after conv3); 153MB < 192MB
  char* tb = (char*)f1; size_t toff = 0;
  auto talloc = [&](size_t bytes)->void*{
    void* p = tb + toff; toff += bytes; toff = (toff + 255) & ~(size_t)255; return p;
  };
  float* xe   = (float*)talloc((size_t)B_*G_*D_*4);
  float* posh = (float*)talloc((size_t)B_*G_*128*4);
  float* posp = (float*)talloc((size_t)B_*G_*D_*4);
  float* xA   = (float*)talloc((size_t)R_*D_*4);
  float* xB   = (float*)talloc((size_t)R_*D_*4);
  float* hbuf = (float*)talloc((size_t)R_*D_*4);
  float* obuf = (float*)talloc((size_t)R_*D_*4);
  float* qkvb = (float*)talloc((size_t)R_*3*D_*4);
  float* ffn  = f3;   // aliases f3 (dead once converted to abuf before conv4)

  // ---- point grouping ----
  fps_kernel<<<B_, 256, 0, stream>>>(pcd, fidx, dists);
  knn_kernel<<<dim3(G_, B_), 256, 0, stream>>>(pcd, fidx, gidx);
  gather_kernel<<<gs(ROWS_), 256, 0, stream>>>(pcd, fidx, gidx, feats, center);

  // ---- mini PointNet ----
  conv1_kernel<<<gs((size_t)ROWS_*128), 256, 0, stream>>>(feats, conv1_w, conv1_b, f1);
  fill_kernel<<<4, 256, 0, stream>>>(stats, 0.0f, 1024);
  bn_stats_kernel<<<ROWS_/256, 256, 0, stream>>>(f1, stats, ROWS_, 128);
  bn_apply_relu_kernel<<<gs((size_t)ROWS_*128), 256, 0, stream>>>(f1, stats, bn1_g, bn1_b, ROWS_, 128);
  convA(stream, f1, abuf, (size_t)ROWS_*128);
  convW(stream, conv2_w, wtbuf, 128, 256);
  gemmL(stream, abuf, nullptr, wtbuf, conv2_b, nullptr, f2, ROWS_, 256, 128, 0, 0, 0);
  rowmax_kernel<<<gs((size_t)B_*G_*256), 256, 0, stream>>>(f2, fg);
  convA(stream, f2, abuf, (size_t)ROWS_*256);
  convA(stream, fg, fgbf, (size_t)B_*G_*256);
  convW(stream, conv3_w, wtbuf, 512, 512);
  gemmL(stream, abuf, fgbf, wtbuf, conv3_b, nullptr, f3, ROWS_, 512, 512, 0, 256, 0); // concat mode
  fill_kernel<<<4, 256, 0, stream>>>(stats, 0.0f, 1024);
  bn_stats_kernel<<<ROWS_/256, 256, 0, stream>>>(f3, stats, ROWS_, 512);
  bn_apply_relu_kernel<<<gs((size_t)ROWS_*512), 256, 0, stream>>>(f3, stats, bn2_g, bn2_b, ROWS_, 512);
  convA(stream, f3, abuf, (size_t)ROWS_*512);
  convW(stream, conv4_w, wtbuf, 512, 512);
  fill_kernel<<<gs((size_t)B_*G_*ENC_), 256, 0, stream>>>(tok, -INFINITY, (size_t)B_*G_*ENC_);
  gemmL(stream, abuf, nullptr, wtbuf, conv4_b, nullptr, tok, ROWS_, 512, 512, 0, 0, K_); // fused max-pool

  // ---- embed + positions + sequence ----
  convA(stream, tok, abuf, (size_t)B_*G_*ENC_);
  convW(stream, e2t_w, wtbuf, 512, 1024);
  gemmL(stream, abuf, nullptr, wtbuf, e2t_b, nullptr, xe, B_*G_, D_, ENC_, 0, 0, 0);
  pos1_kernel<<<gs((size_t)B_*G_*128), 256, 0, stream>>>(center, pos1_w, pos1_b, posh);
  convA(stream, posh, abuf, (size_t)B_*G_*128);
  convW(stream, pos2_w, wtbuf, 128, 1024);
  gemmL(stream, abuf, nullptr, wtbuf, pos2_b, nullptr, posp, B_*G_, D_, 128, 0, 0, 0);
  build_x_kernel<<<gs((size_t)R_*D_), 256, 0, stream>>>(xe, posp, cls_tok, cls_pos, xA);

  // ---- transformer ----
  for (int ll = 0; ll < L_; ++ll){
    ln_kernel<<<R_, 256, 0, stream>>>(xA, hbuf, ln1_g + ll*D_, ln1_b + ll*D_, D_);
    convA(stream, hbuf, abuf, (size_t)R_*D_);
    convW(stream, wqkv + (size_t)ll*D_*3*D_, wtbuf, D_, 3*D_);
    gemmL(stream, abuf, nullptr, wtbuf, bqkv + (size_t)ll*3*D_, nullptr, qkvb, R_, 3*D_, D_, 0, 0, 0);
    convA(stream, qkvb, qkvbf, (size_t)R_*3*D_);
    attn_kernel<<<dim3((S_+15)/16, H_, B_), 32, 0, stream>>>(qkvbf, obuf);
    convA(stream, obuf, abuf, (size_t)R_*D_);
    convW(stream, wo + (size_t)ll*D_*D_, wtbuf, D_, D_);
    gemmL(stream, abuf, nullptr, wtbuf, bo + (size_t)ll*D_, xA, xB, R_, D_, D_, 0, 0, 0); // + residual
    ln_kernel<<<R_, 256, 0, stream>>>(xB, hbuf, ln2_g + ll*D_, ln2_b + ll*D_, D_);
    convA(stream, hbuf, abuf, (size_t)R_*D_);
    convW(stream, w1 + (size_t)ll*D_*DFF_, wtbuf, D_, DFF_);
    gemmL(stream, abuf, nullptr, wtbuf, b1 + (size_t)ll*DFF_, nullptr, ffn, R_, DFF_, D_, 2, 0, 0); // GELU
    convA(stream, ffn, abuf, (size_t)R_*DFF_);
    convW(stream, w2 + (size_t)ll*DFF_*D_, wtbuf, DFF_, D_);
    gemmL(stream, abuf, nullptr, wtbuf, b2 + (size_t)ll*D_, xB, xA, R_, D_, DFF_, 0, 0, 0); // + residual
  }

  // ---- head ----
  final_head_kernel<<<B_, 256, 0, stream>>>(xA, norm_g, norm_b, t2e_w, t2e_b, out);
}
